// BaseTriangleMultiplicativeUpdate_26998164423108
// MI455X (gfx1250) — matrix operand order repaired and checked by reading, not compiled, not run on hardware
//
#include <hip/hip_runtime.h>
#include <hip/hip_bf16.h>

#define N_SEQ 384
#define CZ 128
#define CH 128
#define NROWS (N_SEQ * N_SEQ)   // 147456

typedef __attribute__((ext_vector_type(16))) _Float16 v16h;
typedef __attribute__((ext_vector_type(8)))  _Float16 v8h;
typedef __attribute__((ext_vector_type(8)))  float    v8f;
typedef __attribute__((ext_vector_type(4)))  unsigned int v4u;
typedef __attribute__((ext_vector_type(8)))  int      v8i;
typedef __attribute__((ext_vector_type(4)))  int      v4i;

#if defined(__AMDGCN__) && __has_builtin(__builtin_amdgcn_tensor_load_to_lds) && \
    __has_builtin(__builtin_amdgcn_s_wait_tensorcnt)
#define K_TRI_TDM 1
#else
#define K_TRI_TDM 0
#endif

__device__ __forceinline__ v8f wmma16x16x32(v16h a, v16h b, v8f c) {
  // D = A(16x32 f16) x B(32x16 f16) + C(16x16 f32)
  return __builtin_amdgcn_wmma_f32_16x16x32_f16(false, a, false, b, (short)0, c,
                                                false, false);
}

__device__ __forceinline__ v8f splat8(float x) {
  v8f r;
#pragma unroll
  for (int t = 0; t < 8; ++t) r[t] = x;
  return r;
}

__device__ __forceinline__ float sigm(float x) { return 1.0f / (1.0f + __expf(-x)); }

// A fragment (16 rows x 32 K), row-major f16 source, leading dim ld (elements).
__device__ __forceinline__ v16h load_a_frag(const _Float16* s, int ld, int k0) {
  const int lane = threadIdx.x & 31;
  const int m = lane & 15, hi = lane >> 4;
  const _Float16* p = s + (size_t)m * ld + k0 + hi * 8;
  v8h lo = *(const v8h*)p;          // K = k0+hi*8 .. +7
  v8h hx = *(const v8h*)(p + 16);   // K = k0+16+hi*8 .. +7
  v16h r;
#pragma unroll
  for (int t = 0; t < 8; ++t) { r[t] = lo[t]; r[t + 8] = hx[t]; }
  return r;
}

// B fragment (32 K x 16 cols). Source is transposed bT[col][k] (K-contiguous).
__device__ __forceinline__ v16h load_b_frag(const _Float16* bT, int ld, int k0) {
  const int lane = threadIdx.x & 31;
  const int n = lane & 15, hi = lane >> 4;
  const _Float16* p = bT + (size_t)n * ld + k0 + hi * 8;
  v8h x0 = *(const v8h*)p;
  v8h x1 = *(const v8h*)(p + 16);
  v16h r;
#pragma unroll
  for (int t = 0; t < 8; ++t) { r[t] = x0[t]; r[t + 8] = x1[t]; }
  return r;
}

#if K_TRI_TDM
// Issue one TDM 2D tile load: `rows` rows x 32 f16, row stride `rstride` elems,
// into LDS at lds_addr with 32B padding after each 64B row (-> 48-elem stride).
__device__ __forceinline__ void tdm_load_tile(unsigned int lds_addr,
                                              const _Float16* gptr,
                                              unsigned int rows,
                                              unsigned int rstride) {
  unsigned long long ga = (unsigned long long)(size_t)gptr;
  v4u g0;
  g0[0] = 1u;                                   // count=1, user descriptor
  g0[1] = lds_addr;                             // LDS byte address
  g0[2] = (unsigned int)ga;                     // global_addr[31:0]
  g0[3] = (unsigned int)((ga >> 32) & 0x01FFFFFFu) | 0x80000000u; // addr[56:32] | type=2
  v8i g1;
  // data_size=1 (2B), pad_enable, pad_interval=3 (16 DW = 64B), pad_amount=7 (8 DW = 32B)
  g1[0] = (int)((1u << 16) | (1u << 20) | (3u << 22) | (7u << 25));
  g1[1] = (int)(32u << 16);                     // tensor_dim0[15:0] = 32 @ bits 63:48
  g1[2] = (int)((rows & 0xFFFFu) << 16);        // tensor_dim0 hi=0, tensor_dim1 lo = rows
  g1[3] = (int)(32u << 16);                     // tensor_dim1 hi=0, tile_dim0 = 32
  g1[4] = (int)(rows & 0xFFFFu);                // tile_dim1 = rows, tile_dim2 = 0 (2D)
  g1[5] = (int)rstride;                         // tensor_dim0_stride lo32
  g1[6] = 0;                                    // stride hi, dim1_stride lo (unused, 2D)
  g1[7] = 0;
  v4i gz = {0, 0, 0, 0};
#if __clang_major__ >= 23
  v8i gz8 = {0, 0, 0, 0, 0, 0, 0, 0};
  __builtin_amdgcn_tensor_load_to_lds(g0, g1, gz, gz, gz8, 0);
#else
  __builtin_amdgcn_tensor_load_to_lds(g0, g1, gz, gz, 0);
#endif
}
#endif

// ---------------------------------------------------------------------------
// Kernel 0: weights [k][c] f32 -> wcatT[col][k] f16 (5 fused projections),
// wzT[c][k] f16, biascat[640] f32.
// ---------------------------------------------------------------------------
__global__ __launch_bounds__(256) void k_prep(
    const float* w_ap, const float* w_ag, const float* w_bp, const float* w_bg,
    const float* w_g, const float* w_z, const float* b_ap, const float* b_ag,
    const float* b_bp, const float* b_bg, const float* b_g,
    _Float16* wcatT, _Float16* wzT, float* biascat) {
  int t = blockIdx.x * blockDim.x + threadIdx.x;
  if (t >= 768 * 128) return;
  int col = t >> 7;   // 0..767
  int k = t & 127;
  const float* w; int c;
  if (col < 128)      { w = w_ap; c = col; }
  else if (col < 256) { w = w_ag; c = col - 128; }
  else if (col < 384) { w = w_bp; c = col - 256; }
  else if (col < 512) { w = w_bg; c = col - 384; }
  else if (col < 640) { w = w_g;  c = col - 512; }
  else                { w = w_z;  c = col - 640; }
  float v = w[k * 128 + c];
  if (col < 640) wcatT[(size_t)col * 128 + k] = (_Float16)v;
  else           wzT[(size_t)c * 128 + k] = (_Float16)v;
  if (k == 0 && col < 640) {
    float bv;
    if (col < 128)      bv = b_ap[col];
    else if (col < 256) bv = b_ag[col - 128];
    else if (col < 384) bv = b_bp[col - 256];
    else if (col < 512) bv = b_bg[col - 384];
    else                bv = b_g[col - 512];
    biascat[col] = bv;
  }
}

// ---------------------------------------------------------------------------
// Kernel 1: layernorm(z) + 5 fused projections via WMMA.
// ---------------------------------------------------------------------------
__global__ __launch_bounds__(128) void k_proj(
    const float* __restrict__ z, const float* __restrict__ mask,
    const float* __restrict__ ln_g, const float* __restrict__ ln_b,
    const _Float16* __restrict__ wcatT, const float* __restrict__ biascat,
    _Float16* __restrict__ aT, _Float16* __restrict__ bT,
    _Float16* __restrict__ gbuf) {
  __shared__ _Float16 zns[4][16][136];   // pad 128->136 (272B row, 16B aligned)

  const int wave = threadIdx.x >> 5;
  const int lane = threadIdx.x & 31;
  const int m = lane & 15, hi = lane >> 4;
  const int rowbase = blockIdx.x * 64 + wave * 16;   // flattened (i*384 + k)
  const int iRow = rowbase / N_SEQ;
  const int kz = rowbase - iRow * N_SEQ;             // multiple of 16

  // ---- layernorm over CZ=128: lane pair (m, m+16) splits one row ----
  const float* zr = z + (size_t)(rowbase + m) * CZ + hi * 64;
  float vals[64];
  float s = 0.f, s2 = 0.f;
#pragma unroll
  for (int t = 0; t < 16; ++t) {
    float4 x = ((const float4*)zr)[t];
    vals[4 * t + 0] = x.x; vals[4 * t + 1] = x.y;
    vals[4 * t + 2] = x.z; vals[4 * t + 3] = x.w;
    s  += x.x + x.y + x.z + x.w;
    s2 += x.x * x.x + x.y * x.y + x.z * x.z + x.w * x.w;
  }
  s  += __shfl_xor(s, 16, 32);
  s2 += __shfl_xor(s2, 16, 32);
  const float mean = s * (1.0f / 128.0f);
  const float var = s2 * (1.0f / 128.0f) - mean * mean;
  const float rs = rsqrtf(var + 1e-5f);
#pragma unroll
  for (int t = 0; t < 8; ++t) {
    v8h h;
#pragma unroll
    for (int u = 0; u < 8; ++u) {
      int c = hi * 64 + t * 8 + u;
      h[u] = (_Float16)((vals[t * 8 + u] - mean) * rs * ln_g[c] + ln_b[c]);
    }
    *(v8h*)&zns[wave][m][hi * 64 + t * 8] = h;
  }
  __syncthreads();

  v16h afr[4];
#pragma unroll
  for (int kk = 0; kk < 4; ++kk)
    afr[kk] = load_a_frag(&zns[wave][0][0], 136, kk * 32);

  float mk[8];
#pragma unroll
  for (int v = 0; v < 8; ++v) mk[v] = mask[rowbase + hi * 8 + v];

  const int cn = lane & 15;
  for (int ct = 0; ct < 8; ++ct) {
    const int c0 = ct * 16;
    // a = mask * sigmoid(zn@w_ag) * (zn@w_ap)
    v8f accp = splat8(biascat[c0 + cn]);
    v8f accg = splat8(biascat[128 + c0 + cn]);
#pragma unroll
    for (int kk = 0; kk < 4; ++kk) {
      accp = wmma16x16x32(afr[kk], load_b_frag(wcatT + (size_t)c0 * 128, 128, kk * 32), accp);
      accg = wmma16x16x32(afr[kk], load_b_frag(wcatT + (size_t)(128 + c0) * 128, 128, kk * 32), accg);
    }
    {
      v8h ah;
#pragma unroll
      for (int v = 0; v < 8; ++v)
        ah[v] = (_Float16)(mk[v] * sigm(accg[v]) * accp[v]);
      *(v8h*)(aT + (size_t)(c0 + cn) * NROWS + (size_t)iRow * N_SEQ + kz + hi * 8) = ah;
    }
    // b = mask * sigmoid(zn@w_bg) * (zn@w_bp)
    accp = splat8(biascat[256 + c0 + cn]);
    accg = splat8(biascat[384 + c0 + cn]);
#pragma unroll
    for (int kk = 0; kk < 4; ++kk) {
      accp = wmma16x16x32(afr[kk], load_b_frag(wcatT + (size_t)(256 + c0) * 128, 128, kk * 32), accp);
      accg = wmma16x16x32(afr[kk], load_b_frag(wcatT + (size_t)(384 + c0) * 128, 128, kk * 32), accg);
    }
    {
      v8h bh;
#pragma unroll
      for (int v = 0; v < 8; ++v)
        bh[v] = (_Float16)(mk[v] * sigm(accg[v]) * accp[v]);
      *(v8h*)(bT + (size_t)(c0 + cn) * NROWS + (size_t)iRow * N_SEQ + kz + hi * 8) = bh;
    }
    // g = sigmoid(zn@w_g)
    v8f accG = splat8(biascat[512 + c0 + cn]);
#pragma unroll
    for (int kk = 0; kk < 4; ++kk)
      accG = wmma16x16x32(afr[kk], load_b_frag(wcatT + (size_t)(512 + c0) * 128, 128, kk * 32), accG);
#pragma unroll
    for (int v = 0; v < 8; ++v)
      gbuf[(size_t)(rowbase + hi * 8 + v) * CH + c0 + cn] = (_Float16)sigm(accG[v]);
  }
}

// ---------------------------------------------------------------------------
// Kernel 2: triangle einsum. Per channel c: P_c = A_c x B_c^T (384x384x384).
// 8 waves compute a 128(i) x 64(j) tile. K=32 steps staged into double-
// buffered LDS by the Tensor Data Mover (TDM) when available, overlapping the
// DMA of step k+1 with the WMMAs of step k. TENSORcnt + barriers synchronize.
// ---------------------------------------------------------------------------
__global__ __launch_bounds__(256) void k_tri(const _Float16* __restrict__ aT,
                                             const _Float16* __restrict__ bT,
                                             float* __restrict__ pT) {
  __shared__ _Float16 As[2][128][48];   // 24 KB (48-elem rows: 64B data + 32B pad)
  __shared__ _Float16 Bs[2][64][48];    // 12 KB

  const int c  = blockIdx.z;
  const int i0 = blockIdx.y * 128;
  const int j0 = blockIdx.x * 64;
  const _Float16* Ab = aT + (size_t)c * NROWS + (size_t)i0 * N_SEQ;
  const _Float16* Bb = bT + (size_t)c * NROWS + (size_t)j0 * N_SEQ;

  const int tid = threadIdx.x;
  const int wave = tid >> 5, lane = tid & 31;
  const int wy = wave >> 1, wx = wave & 1;    // 4 x 2 wave grid
  const int m = lane & 15, hi = lane >> 4;

  v8f acc[2][2];
#pragma unroll
  for (int r = 0; r < 2; ++r)
#pragma unroll
    for (int ss = 0; ss < 2; ++ss) acc[r][ss] = splat8(0.f);

#if K_TRI_TDM
  const unsigned int ldsA[2] = {(unsigned int)(size_t)&As[0][0][0],
                                (unsigned int)(size_t)&As[1][0][0]};
  const unsigned int ldsB[2] = {(unsigned int)(size_t)&Bs[0][0][0],
                                (unsigned int)(size_t)&Bs[1][0][0]};
  const bool issuer = (tid < 32);   // wave 0 drives the TDM
  if (issuer) {
    tdm_load_tile(ldsA[0], Ab, 128u, (unsigned int)N_SEQ);
    tdm_load_tile(ldsB[0], Bb, 64u, (unsigned int)N_SEQ);
  }
  for (int it = 0; it < 12; ++it) {
    const int buf = it & 1;
    __syncthreads();   // everyone done computing on buf^1 (iter it-1)
    if (it + 1 < 12) {
      if (issuer) {
        tdm_load_tile(ldsA[buf ^ 1], Ab + (it + 1) * 32, 128u, (unsigned int)N_SEQ);
        tdm_load_tile(ldsB[buf ^ 1], Bb + (it + 1) * 32, 64u, (unsigned int)N_SEQ);
      }
      if (issuer) __builtin_amdgcn_s_wait_tensorcnt(2);  // iter it's pair done
    } else {
      if (issuer) __builtin_amdgcn_s_wait_tensorcnt(0);
    }
    __syncthreads();   // publish TDM-written LDS to all waves

    v16h af[2], bf[2];
#pragma unroll
    for (int r = 0; r < 2; ++r)
      af[r] = load_a_frag(&As[buf][wy * 32 + r * 16][0], 48, 0);
#pragma unroll
    for (int ss = 0; ss < 2; ++ss)
      bf[ss] = load_b_frag(&Bs[buf][wx * 32 + ss * 16][0], 48, 0);
#pragma unroll
    for (int r = 0; r < 2; ++r)
#pragma unroll
      for (int ss = 0; ss < 2; ++ss)
        acc[r][ss] = wmma16x16x32(af[r], bf[ss], acc[r][ss]);
  }
#else
  for (int k0 = 0; k0 < N_SEQ; k0 += 32) {
    __syncthreads();
#pragma unroll
    for (int rep = 0; rep < 2; ++rep) {
      int idx = tid * 2 + rep;            // 0..511
      int row = idx >> 2, ch = idx & 3;
      *(v8h*)&As[0][row][ch * 8] = *(const v8h*)(Ab + (size_t)row * N_SEQ + k0 + ch * 8);
    }
    {
      int row = tid >> 2, ch = tid & 3;   // 0..63
      *(v8h*)&Bs[0][row][ch * 8] = *(const v8h*)(Bb + (size_t)row * N_SEQ + k0 + ch * 8);
    }
    __syncthreads();

    v16h af[2], bf[2];
#pragma unroll
    for (int r = 0; r < 2; ++r) af[r] = load_a_frag(&As[0][wy * 32 + r * 16][0], 48, 0);
#pragma unroll
    for (int ss = 0; ss < 2; ++ss) bf[ss] = load_b_frag(&Bs[0][wx * 32 + ss * 16][0], 48, 0);
#pragma unroll
    for (int r = 0; r < 2; ++r)
#pragma unroll
      for (int ss = 0; ss < 2; ++ss)
        acc[r][ss] = wmma16x16x32(af[r], bf[ss], acc[r][ss]);
  }
#endif

  // store pT[c][i][j] (coalesced over j via lanes 0..15)
#pragma unroll
  for (int r = 0; r < 2; ++r)
#pragma unroll
    for (int ss = 0; ss < 2; ++ss) {
      const int jj = j0 + wx * 32 + ss * 16 + m;
#pragma unroll
      for (int v = 0; v < 8; ++v) {
        const int ii = i0 + wy * 32 + r * 16 + v + hi * 8;
        pT[(size_t)c * NROWS + (size_t)ii * N_SEQ + jj] = acc[r][ss][v];
      }
    }
}

// ---------------------------------------------------------------------------
// Kernel 3: out = (layernorm_c(p) @ w_z + b_z) * g.
// ---------------------------------------------------------------------------
__global__ __launch_bounds__(256) void k_final(
    const float* __restrict__ pT, const _Float16* __restrict__ gbuf,
    const _Float16* __restrict__ wzT, const float* __restrict__ bz,
    const float* __restrict__ ln_g, const float* __restrict__ ln_b,
    float* __restrict__ out) {
  __shared__ _Float16 znp[8][16][136];

  const int wave = threadIdx.x >> 5, lane = threadIdx.x & 31;
  const int m = lane & 15, hi = lane >> 4;
  const size_t ijbase = (size_t)blockIdx.x * 128 + wave * 16;
  const size_t ij = ijbase + m;

  float vals[64];
  float s = 0.f, s2 = 0.f;
#pragma unroll 8
  for (int t = 0; t < 64; ++t) {
    float x = pT[(size_t)(hi * 64 + t) * NROWS + ij];
    vals[t] = x; s += x; s2 += x * x;
  }
  s  += __shfl_xor(s, 16, 32);
  s2 += __shfl_xor(s2, 16, 32);
  const float mean = s * (1.0f / 128.0f);
  const float var = s2 * (1.0f / 128.0f) - mean * mean;
  const float rs = rsqrtf(var + 1e-5f);
#pragma unroll
  for (int t = 0; t < 8; ++t) {
    v8h h;
#pragma unroll
    for (int u = 0; u < 8; ++u) {
      int cch = hi * 64 + t * 8 + u;
      h[u] = (_Float16)((vals[t * 8 + u] - mean) * rs * ln_g[cch] + ln_b[cch]);
    }
    *(v8h*)&znp[wave][m][hi * 64 + t * 8] = h;
  }
  __syncthreads();

  v16h afr[4];
#pragma unroll
  for (int kk = 0; kk < 4; ++kk)
    afr[kk] = load_a_frag(&znp[wave][0][0], 136, kk * 32);

  const int cn = lane & 15;
  for (int ct = 0; ct < 8; ++ct) {
    const int c0 = ct * 16;
    v8f acc = splat8(bz[c0 + cn]);
#pragma unroll
    for (int kk = 0; kk < 4; ++kk)
      acc = wmma16x16x32(afr[kk], load_b_frag(wzT + (size_t)c0 * 128, 128, kk * 32), acc);
#pragma unroll
    for (int v = 0; v < 8; ++v) {
      const size_t r = ijbase + hi * 8 + v;
      const float gv = (float)gbuf[r * CH + c0 + cn];
      out[r * CH + c0 + cn] = acc[v] * gv;
    }
  }
}

// ---------------------------------------------------------------------------
extern "C" void kernel_launch(void* const* d_in, const int* in_sizes, int n_in,
                              void* d_out, int out_size, void* d_ws, size_t ws_size,
                              hipStream_t stream) {
  const float* z       = (const float*)d_in[0];
  const float* mask    = (const float*)d_in[1];
  const float* w_ap    = (const float*)d_in[2];
  const float* b_ap    = (const float*)d_in[3];
  const float* w_ag    = (const float*)d_in[4];
  const float* b_ag    = (const float*)d_in[5];
  const float* w_bp    = (const float*)d_in[6];
  const float* b_bp    = (const float*)d_in[7];
  const float* w_bg    = (const float*)d_in[8];
  const float* b_bg    = (const float*)d_in[9];
  const float* w_g     = (const float*)d_in[10];
  const float* b_g     = (const float*)d_in[11];
  const float* w_z     = (const float*)d_in[12];
  const float* b_z     = (const float*)d_in[13];
  const float* ln_in_g = (const float*)d_in[14];
  const float* ln_in_b = (const float*)d_in[15];
  const float* ln_out_g= (const float*)d_in[16];
  const float* ln_out_b= (const float*)d_in[17];

  char* ws = (char*)d_ws;
  size_t off = 0;
  auto take = [&](size_t bytes) -> char* {
    char* p = ws + off;
    off = (off + bytes + 255) & ~(size_t)255;
    return p;
  };
  _Float16* wcatT  = (_Float16*)take((size_t)640 * 128 * 2);
  _Float16* wzT    = (_Float16*)take((size_t)128 * 128 * 2);
  float*    biascat= (float*)take((size_t)640 * 4);
  _Float16* aT     = (_Float16*)take((size_t)CH * NROWS * 2);
  _Float16* bTm    = (_Float16*)take((size_t)CH * NROWS * 2);
  _Float16* gbuf   = (_Float16*)take((size_t)NROWS * CH * 2);
  float*    pT     = (float*)take((size_t)CH * NROWS * 4);

  k_prep<<<dim3((768 * 128 + 255) / 256), dim3(256), 0, stream>>>(
      w_ap, w_ag, w_bp, w_bg, w_g, w_z, b_ap, b_ag, b_bp, b_bg, b_g,
      wcatT, wzT, biascat);

  k_proj<<<dim3(NROWS / 64), dim3(128), 0, stream>>>(
      z, mask, ln_in_g, ln_in_b, wcatT, biascat, aT, bTm, gbuf);

  k_tri<<<dim3(N_SEQ / 64, N_SEQ / 128, CH), dim3(256), 0, stream>>>(aT, bTm, pT);

  k_final<<<dim3(NROWS / 128), dim3(256), 0, stream>>>(
      pT, gbuf, wzT, b_z, ln_out_g, ln_out_b, (float*)d_out);
}
